// GNNGL_PPI_60533269069967
// MI455X (gfx1250) — compile-verified, matrix operand order checked by reading, not run on hardware
//
#include <hip/hip_runtime.h>

// ---------------------------------------------------------------------------
// Types for CDNA5 WMMA (wave32): 16x16x32 bf16 -> f32 accumulate
// ---------------------------------------------------------------------------
typedef __attribute__((ext_vector_type(16))) __bf16 v16bf;
typedef __attribute__((ext_vector_type(8)))  float  v8f;

#define D512 512

union FragU { v16bf v; uint4 u[2]; };

// ---------------------------------------------------------------------------
// CDNA5 async DMA helpers (inline asm: portable across ROCm7.2 / clang-23,
// the clang builtins have toolchain-dependent arity).
// global_load_async_to_lds_b128: VDST = 32-bit LDS byte address VGPR,
// VADDR = 64-bit global address; tracked by ASYNCcnt.
// ---------------------------------------------------------------------------
__device__ __forceinline__ unsigned lds_addr(const void* p)
{
    // generic pointer to LDS: low 32 bits are the LDS byte offset
    return (unsigned)(size_t)p;
}

__device__ __forceinline__ void async_load_b128(const void* gaddr, unsigned lds)
{
    asm volatile("global_load_async_to_lds_b128 %0, %1, off"
                 :: "v"(lds), "v"(gaddr)
                 : "memory");
}

template <int N>
__device__ __forceinline__ void wait_asynccnt()
{
    asm volatile("s_wait_asynccnt %0" :: "i"(N) : "memory");
}

// ---------------------------------------------------------------------------
// Main tiled GEMM:  out[M x 512] = A[M x 512](bf16) @ W[512 x 512]
// Wt is pre-transposed bf16 [Nout][K].  Optional bias / residual / relu,
// optional fp32 and bf16 outputs.
// Block = 256 threads (8 waves). Block tile 128(M) x 64(N).
// Each wave computes a 32x32 output patch = 2x2 WMMA 16x16 tiles.
// K pipeline: double-buffered LDS, async global->LDS DMA for tile k+1
// overlapped with WMMA on tile k; s_wait_asynccnt 3 retires the older tile.
// ---------------------------------------------------------------------------
__global__ __launch_bounds__(256)
void gemm_bf16_wmma(const __bf16* __restrict__ A, const __bf16* __restrict__ Wt,
                    const float* __restrict__ bias, const float* __restrict__ resid,
                    float* __restrict__ outF, __bf16* __restrict__ outB,
                    int M, int relu)
{
    // padded stride 40 bf16 (80B): row offsets l*20 mod 64 banks -> conflict-free
    __shared__ __align__(16) __bf16 As[2][128 * 40];
    __shared__ __align__(16) __bf16 Bs[2][64 * 40];

    const int tid   = threadIdx.x;
    const int lane  = tid & 31;
    const int wv    = tid >> 5;
    const int wm    = (wv >> 1) << 5;   // 0,32,64,96
    const int wn    = (wv & 1) << 5;    // 0,32
    const int l15   = lane & 15;
    const int half  = lane >> 4;        // 0: K 0-7/16-23, 1: K 8-15/24-31
    const int mbase = blockIdx.x * 128;
    const int nbase = blockIdx.y * 64;

    // Per-thread staging slots (fixed for the whole kernel)
    const int arow0 = (tid)       >> 2, aq0 = (tid)       & 3;
    const int arow1 = (tid + 256) >> 2, aq1 = (tid + 256) & 3;
    const int brow  = tid >> 2,         bq  = tid & 3;
    int ag0 = mbase + arow0; if (ag0 >= M) ag0 = M - 1;   // clamp; masked at epilogue
    int ag1 = mbase + arow1; if (ag1 >= M) ag1 = M - 1;
    const int bg = nbase + brow;

    // Issue the 3 async B128 DMA ops for K-tile k0 into buffer `buf`
    auto stage = [&](int k0, int buf) {
        async_load_b128(&A[(size_t)ag0 * D512 + k0 + aq0 * 8],
                        lds_addr(&As[buf][arow0 * 40 + aq0 * 8]));
        async_load_b128(&A[(size_t)ag1 * D512 + k0 + aq1 * 8],
                        lds_addr(&As[buf][arow1 * 40 + aq1 * 8]));
        async_load_b128(&Wt[(size_t)bg * D512 + k0 + bq * 8],
                        lds_addr(&Bs[buf][brow * 40 + bq * 8]));
        if (k0 + 32 < D512)   // keep the L2 prefetch path warm one tile further out
            __builtin_prefetch(&A[(size_t)ag0 * D512 + k0 + 32 + aq0 * 8], 0, 1);
    };

    v8f acc00 = {}, acc01 = {}, acc10 = {}, acc11 = {};

    stage(0, 0);
    for (int ks = 0; ks < 16; ++ks) {
        const int p = ks & 1;
        if (ks + 1 < 16) {
            stage((ks + 1) * 32, p ^ 1);
            wait_asynccnt<3>();   // oldest 3 (this tile) done; next tile in flight
        } else {
            wait_asynccnt<0>();
        }
        __syncthreads();

        // ---- build fragments per ISA 16-bit A/B layouts ----
        FragU a0, a1, b0, b1;
        {
            int r0 = wm + l15;
            a0.u[0] = *(const uint4*)(&As[p][r0 * 40 + half * 8]);
            a0.u[1] = *(const uint4*)(&As[p][r0 * 40 + 16 + half * 8]);
            int r1 = wm + 16 + l15;
            a1.u[0] = *(const uint4*)(&As[p][r1 * 40 + half * 8]);
            a1.u[1] = *(const uint4*)(&As[p][r1 * 40 + 16 + half * 8]);
            int c0 = wn + l15;
            b0.u[0] = *(const uint4*)(&Bs[p][c0 * 40 + half * 8]);
            b0.u[1] = *(const uint4*)(&Bs[p][c0 * 40 + 16 + half * 8]);
            int c1 = wn + 16 + l15;
            b1.u[0] = *(const uint4*)(&Bs[p][c1 * 40 + half * 8]);
            b1.u[1] = *(const uint4*)(&Bs[p][c1 * 40 + 16 + half * 8]);
        }

        acc00 = __builtin_amdgcn_wmma_f32_16x16x32_bf16(false, a0.v, false, b0.v, (short)0, acc00, false, false);
        acc01 = __builtin_amdgcn_wmma_f32_16x16x32_bf16(false, a0.v, false, b1.v, (short)0, acc01, false, false);
        acc10 = __builtin_amdgcn_wmma_f32_16x16x32_bf16(false, a1.v, false, b0.v, (short)0, acc10, false, false);
        acc11 = __builtin_amdgcn_wmma_f32_16x16x32_bf16(false, a1.v, false, b1.v, (short)0, acc11, false, false);

        __syncthreads();  // all waves done reading buf p before it is re-filled
    }

    // ---- epilogue: D layout = VGPR r, lanes<16 -> M=r, lanes>=16 -> M=r+8 ----
    auto epi = [&](v8f a, int i, int j) {
#pragma unroll
        for (int r = 0; r < 8; ++r) {
            int row = mbase + wm + i * 16 + half * 8 + r;
            int col = nbase + wn + j * 16 + l15;
            if (row < M) {
                float v = a[r];
                if (bias)  v += bias[col];
                if (resid) v += resid[(size_t)row * D512 + col];
                if (relu)  v = v > 0.0f ? v : 0.0f;
                if (outF)  outF[(size_t)row * D512 + col] = v;
                if (outB)  outB[(size_t)row * D512 + col] = (__bf16)v;
            }
        }
    };
    epi(acc00, 0, 0); epi(acc01, 0, 1); epi(acc10, 1, 0); epi(acc11, 1, 1);
}

// ---------------------------------------------------------------------------
// Weight fp32 [K][N] -> bf16 transposed [N][K]
// ---------------------------------------------------------------------------
__global__ void wtrans_bf16(const float* __restrict__ W, __bf16* __restrict__ Wt)
{
    int idx = blockIdx.x * 256 + threadIdx.x;
    if (idx < D512 * D512) {
        int k = idx >> 9;
        int n = idx & 511;
        Wt[(size_t)n * D512 + k] = (__bf16)W[(size_t)k * D512 + n];
    }
}

// fp32 -> bf16 elementwise
__global__ void cvt_bf16(const float* __restrict__ in, __bf16* __restrict__ out, int n)
{
    int i = blockIdx.x * 256 + threadIdx.x;
    if (i < n) out[i] = (__bf16)in[i];
}

// out = in * scale, scale = 1 + *eps (or 1 if eps == nullptr)
__global__ void copy_scale(const float* __restrict__ in, float* __restrict__ out,
                           const float* __restrict__ eps, int n)
{
    int i = blockIdx.x * 256 + threadIdx.x;
    float s = eps ? (1.0f + *eps) : 1.0f;
    if (i < n) out[i] = in[i] * s;
}

// ---------------------------------------------------------------------------
// Segment-sum:  out[dst[e]][:] += in[src[e]][:]   (global_atomic_add_f32)
// One block (128 threads, float4 per thread) per edge.
// ---------------------------------------------------------------------------
__global__ __launch_bounds__(128)
void scatter_add(const float* __restrict__ in, const int* __restrict__ src,
                 const int* __restrict__ dst, float* __restrict__ out)
{
    int e = blockIdx.x;
    int s = src[e];
    int d = dst[e];
    int c = threadIdx.x * 4;
    float4 v = *(const float4*)(&in[(size_t)s * D512 + c]);
    float* o = &out[(size_t)d * D512 + c];
    atomicAdd(o + 0, v.x);
    atomicAdd(o + 1, v.y);
    atomicAdd(o + 2, v.z);
    atomicAdd(o + 3, v.w);
}

// ---------------------------------------------------------------------------
// BatchNorm: column sum/sumsq partials -> atomics into stats[0..511]/[512..1023]
// ---------------------------------------------------------------------------
__global__ __launch_bounds__(256)
void bn_stats(const float* __restrict__ X, float* __restrict__ stats, int M)
{
    int c  = threadIdx.x;            // columns c and c+256
    int r0 = blockIdx.x * 128;
    int r1 = r0 + 128; if (r1 > M) r1 = M;
    float s0 = 0.f, s1 = 0.f, q0 = 0.f, q1 = 0.f;
    for (int r = r0; r < r1; ++r) {
        float a = X[(size_t)r * D512 + c];
        float b = X[(size_t)r * D512 + c + 256];
        s0 += a; q0 += a * a;
        s1 += b; q1 += b * b;
    }
    atomicAdd(&stats[c], s0);
    atomicAdd(&stats[c + 256], s1);
    atomicAdd(&stats[512 + c], q0);
    atomicAdd(&stats[512 + c + 256], q1);
}

__global__ void bn_finalize(float* __restrict__ stats, int M)
{
    int c = blockIdx.x * 256 + threadIdx.x;
    if (c < D512) {
        float inv  = 1.0f / (float)M;
        float mean = stats[c] * inv;
        float var  = stats[512 + c] * inv - mean * mean;
        stats[c]       = mean;
        stats[512 + c] = rsqrtf(var + 1e-5f);
    }
}

__global__ void bn_apply(const float* __restrict__ X, const float* __restrict__ stats,
                         const float* __restrict__ g, const float* __restrict__ b,
                         float* __restrict__ outF, __bf16* __restrict__ outB,
                         int total, int relu)
{
    int i = blockIdx.x * 256 + threadIdx.x;
    if (i < total) {
        int c   = i & 511;
        float v = (X[i] - stats[c]) * stats[512 + c] * g[c] + b[c];
        if (relu) v = v > 0.0f ? v : 0.0f;
        if (outF) outF[i] = v;
        if (outB) outB[i] = (__bf16)v;
    }
}

// ---------------------------------------------------------------------------
// Final:  pair = h[n0]*h[n1];  out[t] = pair @ fc2_W[512x7] + fc2_b
// One 64-thread block per training edge.
// ---------------------------------------------------------------------------
__global__ __launch_bounds__(64)
void pair_fc2(const float* __restrict__ h, const int* __restrict__ ei,
              const int* __restrict__ teid, const float* __restrict__ W,
              const float* __restrict__ b, float* __restrict__ out, int E)
{
    int t  = blockIdx.x;
    int id = teid[t];
    int n0 = ei[id];
    int n1 = ei[E + id];

    float acc[7] = {0.f, 0.f, 0.f, 0.f, 0.f, 0.f, 0.f};
    for (int i = threadIdx.x; i < D512; i += 64) {
        float p = h[(size_t)n0 * D512 + i] * h[(size_t)n1 * D512 + i];
#pragma unroll
        for (int c = 0; c < 7; ++c) acc[c] += p * W[i * 7 + c];
    }

    __shared__ float red[64][8];
#pragma unroll
    for (int c = 0; c < 7; ++c) red[threadIdx.x][c] = acc[c];
    __syncthreads();
    if (threadIdx.x < 7) {
        float s = 0.f;
        for (int j = 0; j < 64; ++j) s += red[j][threadIdx.x];
        out[(size_t)t * 7 + threadIdx.x] = s + b[threadIdx.x];
    }
}

// ---------------------------------------------------------------------------
// Launcher
// ---------------------------------------------------------------------------
extern "C" void kernel_launch(void* const* d_in, const int* in_sizes, int n_in,
                              void* d_out, int out_size, void* d_ws, size_t ws_size,
                              hipStream_t stream)
{
    const float* x      = (const float*)d_in[0];
    const float* graph  = (const float*)d_in[1];
    const int*   ei     = (const int*)d_in[2];    // [2][E]
    const int*   teid   = (const int*)d_in[3];    // [T]
    const float* W_sub  = (const float*)d_in[4];
    const float* bn1_g  = (const float*)d_in[5];
    const float* bn1_b  = (const float*)d_in[6];
    const float* fcx_W  = (const float*)d_in[7];
    const float* fcx_b  = (const float*)d_in[8];
    const float* eps1   = (const float*)d_in[9];
    const float* g_W1   = (const float*)d_in[10];
    const float* g_b1   = (const float*)d_in[11];
    const float* g_W2   = (const float*)d_in[12];
    const float* g_b2   = (const float*)d_in[13];
    const float* gbn_g  = (const float*)d_in[14];
    const float* gbn_b  = (const float*)d_in[15];
    const float* lin1_W = (const float*)d_in[16];
    const float* lin1_b = (const float*)d_in[17];
    const float* lin2_W = (const float*)d_in[18];
    const float* lin2_b = (const float*)d_in[19];
    const float* fc2_W  = (const float*)d_in[20];
    const float* fc2_b  = (const float*)d_in[21];

    const int N = in_sizes[0] / D512;   // 50000
    const int E = in_sizes[2] / 2;      // 200000
    const int T = in_sizes[3];          // 100000
    const int ND = N * D512;            // 25.6M elems

    // ---- workspace carve-up ----
    char* ws = (char*)d_ws;
    size_t off = 0;
    float* F0 = (float*)(ws + off);  off += (size_t)ND * 4;         // sub_x / agg_s
    float* F1 = (float*)(ws + off);  off += (size_t)ND * 4;         // h (fp32)
    float* F2 = (float*)(ws + off);  off += (size_t)ND * 4;         // agg scratch
    __bf16* Bact  = (__bf16*)(ws + off); off += (size_t)ND * 2;     // bf16 ping
    __bf16* Bact2 = (__bf16*)(ws + off); off += (size_t)ND * 2;     // bf16 pong
    __bf16* Bw    = (__bf16*)(ws + off); off += (size_t)6 * D512 * D512 * 2;
    float* stats  = (float*)(ws + off);  off += 1024 * 4;

    const int EL = (ND + 255) / 256;          // elementwise grid
    const int WG = (D512 * D512 + 255) / 256; // weight grid
    dim3 ggrid((N + 127) / 128, 8);

    // ---- 0. weights -> bf16 transposed ----
    const float* wsrc[6] = { W_sub, fcx_W, g_W1, g_W2, lin1_W, lin2_W };
    for (int w = 0; w < 6; ++w)
        wtrans_bf16<<<WG, 256, 0, stream>>>(wsrc[w], Bw + (size_t)w * D512 * D512);

    // ---- 1. agg_s = graph + segsum(graph[src] -> dst) ----
    copy_scale<<<EL, 256, 0, stream>>>(graph, F0, nullptr, ND);
    scatter_add<<<E, 128, 0, stream>>>(graph, ei, ei + E, F0);
    cvt_bf16<<<EL, 256, 0, stream>>>(F0, Bact, ND);

    // ---- 2. sub_x_pre = agg_s @ W_sub + graph ----
    gemm_bf16_wmma<<<ggrid, 256, 0, stream>>>(Bact, Bw + 0 * (size_t)D512 * D512,
                                              nullptr, graph, F1, nullptr, N, 0);

    // ---- 3. sub_x = relu(BN(sub_x_pre)) -> F0 ----
    hipMemsetAsync(stats, 0, 1024 * sizeof(float), stream);
    bn_stats<<<(N + 127) / 128, 256, 0, stream>>>(F1, stats, N);
    bn_finalize<<<2, 256, 0, stream>>>(stats, N);
    bn_apply<<<EL, 256, 0, stream>>>(F1, stats, bn1_g, bn1_b, F0, nullptr, ND, 1);

    // ---- 4. h = x @ fcx_W + fcx_b ----
    cvt_bf16<<<EL, 256, 0, stream>>>(x, Bact, ND);
    gemm_bf16_wmma<<<ggrid, 256, 0, stream>>>(Bact, Bw + 1 * (size_t)D512 * D512,
                                              fcx_b, nullptr, F1, nullptr, N, 0);

    // ---- 5. agg = (1+eps1)*h + segsum(h[src] -> dst) ----
    copy_scale<<<EL, 256, 0, stream>>>(F1, F2, eps1, ND);
    scatter_add<<<E, 128, 0, stream>>>(F1, ei, ei + E, F2);
    cvt_bf16<<<EL, 256, 0, stream>>>(F2, Bact, ND);

    // ---- 6. h = relu(agg @ g_W1 + g_b1) -> bf16 only ----
    gemm_bf16_wmma<<<ggrid, 256, 0, stream>>>(Bact, Bw + 2 * (size_t)D512 * D512,
                                              g_b1, nullptr, nullptr, Bact2, N, 1);

    // ---- 7. h = relu(h @ g_W2 + g_b2) -> fp32 for BN ----
    gemm_bf16_wmma<<<ggrid, 256, 0, stream>>>(Bact2, Bw + 3 * (size_t)D512 * D512,
                                              g_b2, nullptr, F1, nullptr, N, 1);

    // ---- 8. h = BN(h) -> bf16 ----
    hipMemsetAsync(stats, 0, 1024 * sizeof(float), stream);
    bn_stats<<<(N + 127) / 128, 256, 0, stream>>>(F1, stats, N);
    bn_finalize<<<2, 256, 0, stream>>>(stats, N);
    bn_apply<<<EL, 256, 0, stream>>>(F1, stats, gbn_g, gbn_b, nullptr, Bact, ND, 0);

    // ---- 9. h = relu(h @ lin1_W + lin1_b) -> bf16 ----
    gemm_bf16_wmma<<<ggrid, 256, 0, stream>>>(Bact, Bw + 4 * (size_t)D512 * D512,
                                              lin1_b, nullptr, nullptr, Bact2, N, 1);

    // ---- 10. h = h @ lin2_W + lin2_b + sub_x -> F1 ----
    gemm_bf16_wmma<<<ggrid, 256, 0, stream>>>(Bact2, Bw + 5 * (size_t)D512 * D512,
                                              lin2_b, F0, F1, nullptr, N, 0);

    // ---- 11. pair gather + fc2 projection ----
    pair_fc2<<<T, 64, 0, stream>>>(F1, ei, teid, fc2_W, fc2_b, (float*)d_out, E);
}
